// LSTM2d_77360950936299
// MI455X (gfx1250) — compile-verified, matrix operand order checked by reading, not run on hardware
//
#include <hip/hip_runtime.h>
#include <hip/hip_bf16.h>

typedef __attribute__((ext_vector_type(16))) __bf16 v16bf;
typedef __attribute__((ext_vector_type(8)))  __bf16 v8bf;
typedef __attribute__((ext_vector_type(8)))  float  v8f;

#define BSZ   64
#define LIN_  32
#define LOUT_ 32
#define E_    256
#define DENC_ 256
#define DD    512
#define VOUT_ 32000

__device__ __forceinline__ float sigm_(float x) { return 1.0f / (1.0f + __expf(-x)); }

// ---------------------------------------------------------------------------
// WMMA fragment loaders (wave32, v_wmma_f32_16x16x32_bf16)
// A: 16x32 (MxK) bf16. lane<16: row=lane, K 0..7 & 16..23; lane>=16: row=lane-16,
//    K 8..15 & 24..31  -> two contiguous 8-element (16B) chunks per lane.
// B: packed layout: for n-tile nt (16 cols) and k-tile kt (32 rows), 32 lanes x
//    16 bf16 contiguous, mirroring A's K mapping with N in place of M.
// C/D: VGPR r, lane: n = n0 + lane%16 ; m = m0 + (lane/16)*8 + r.
// ---------------------------------------------------------------------------
__device__ __forceinline__ v16bf load_a_frag(const __bf16* __restrict__ A, int lda,
                                             int m0, int kk, int lane) {
    int row   = m0 + (lane & 15);
    int kbase = (lane >> 4) * 8;
    const __bf16* p = A + (size_t)row * lda + kk + kbase;
    v8bf lo = *(const v8bf*)p;
    v8bf hi = *(const v8bf*)(p + 16);
    v16bf a;
#pragma unroll
    for (int e = 0; e < 8; ++e) { a[e] = lo[e]; a[8 + e] = hi[e]; }
    return a;
}

__device__ __forceinline__ const __bf16* b_frag_ptr(const __bf16* __restrict__ Bp,
                                                    int Kdim, int n, int kk, int lane) {
    return Bp + ((size_t)(n >> 4) * (size_t)(Kdim >> 5) + (size_t)(kk >> 5)) * 512 +
           (size_t)lane * 16;
}

// Fragment set for a 32(M)x64(N)x32(K) wave step: 2 A frags + 4 B frags.
struct FragSet {
    v16bf a0, a1, b0, b1, b2, b3;
};

__device__ __forceinline__ FragSet load_frags(const __bf16* __restrict__ A, int lda,
                                              int m0, const __bf16* __restrict__ Bp,
                                              int Kdim, int n0, int kk, int lane) {
    FragSet f;
    f.a0 = load_a_frag(A, lda, m0, kk, lane);
    f.a1 = load_a_frag(A, lda, m0 + 16, kk, lane);
    f.b0 = *(const v16bf*)b_frag_ptr(Bp, Kdim, n0, kk, lane);
    f.b1 = *(const v16bf*)b_frag_ptr(Bp, Kdim, n0 + 16, kk, lane);
    f.b2 = *(const v16bf*)b_frag_ptr(Bp, Kdim, n0 + 32, kk, lane);
    f.b3 = *(const v16bf*)b_frag_ptr(Bp, Kdim, n0 + 48, kk, lane);
    return f;
}

#define WMMA_BF16(A_, B_, C_) \
    __builtin_amdgcn_wmma_f32_16x16x32_bf16(false, (A_), false, (B_), (short)0, (C_), false, false)

__device__ __forceinline__ void do_wmma(const FragSet& f, v8f (&acc)[2][4]) {
    acc[0][0] = WMMA_BF16(f.a0, f.b0, acc[0][0]);
    acc[1][0] = WMMA_BF16(f.a1, f.b0, acc[1][0]);
    acc[0][1] = WMMA_BF16(f.a0, f.b1, acc[0][1]);
    acc[1][1] = WMMA_BF16(f.a1, f.b1, acc[1][1]);
    acc[0][2] = WMMA_BF16(f.a0, f.b2, acc[0][2]);
    acc[1][2] = WMMA_BF16(f.a1, f.b2, acc[1][2]);
    acc[0][3] = WMMA_BF16(f.a0, f.b3, acc[0][3]);
    acc[1][3] = WMMA_BF16(f.a1, f.b3, acc[1][3]);
}

__device__ __forceinline__ void store_tile(v8f (&acc)[2][4], float* __restrict__ C,
                                           int ldc, int m0, int n0,
                                           const float* __restrict__ bias, int lane) {
    int nl = lane & 15;
    int mh = (lane >> 4) * 8;
#pragma unroll
    for (int h = 0; h < 2; ++h) {
#pragma unroll
        for (int s = 0; s < 4; ++s) {
            int n = n0 + s * 16 + nl;
            float bv = bias ? bias[n] : 0.0f;
#pragma unroll
            for (int r = 0; r < 8; ++r)
                C[(size_t)(m0 + h * 16 + mh + r) * ldc + n] = acc[h][s][r] + bv;
        }
    }
}

// ---------------------------------------------------------------------------
// Generic bf16 GEMM: C(MxN,f32) = A(MxK,bf16 row-major) * Bpacked(KxN) [+ bias]
// block = 256 threads = 8 waves (2M x 4N); wave tile 32x64; block tile 64x256.
// Requires M%64==0, N%256==0, K%64==0 (true for all shapes here).
// Double-buffered: loads for slab kk+32 are issued before WMMAs of slab kk.
// __launch_bounds__(256, 2): we only ever need 2 waves/SIMD (8-wave block on a
// WGP), so give the compiler the VGPR headroom for both fragment buffers ->
// no scratch spills in the hot loop.
// ---------------------------------------------------------------------------
__global__ void __launch_bounds__(256, 2)
gemm_bf16(const __bf16* __restrict__ A, int lda,
          const __bf16* __restrict__ Bp,
          const float* __restrict__ bias,
          float* __restrict__ C, int ldc, int Kdim) {
    int lane = threadIdx.x & 31;
    int wave = threadIdx.x >> 5;
    int m0 = blockIdx.y * 64 + (wave & 1) * 32;
    int n0 = blockIdx.x * 256 + (wave >> 1) * 64;
    v8f acc[2][4] = {};

    FragSet f0 = load_frags(A, lda, m0, Bp, Kdim, n0, 0, lane);
    for (int kk = 0; kk < Kdim; kk += 64) {
        FragSet f1 = load_frags(A, lda, m0, Bp, Kdim, n0, kk + 32, lane);
        do_wmma(f0, acc);
        if (kk + 64 < Kdim) {
            f0 = load_frags(A, lda, m0, Bp, Kdim, n0, kk + 64, lane);
            if (kk + 128 < Kdim)
                __builtin_prefetch(b_frag_ptr(Bp, Kdim, n0, kk + 128, lane), 0, 1);
        }
        do_wmma(f1, acc);
    }
    store_tile(acc, C, ldc, m0, n0, bias, lane);
}

// ---------------------------------------------------------------------------
// Decoder recurrent GEMM over one anti-diagonal: for cell (i, j=d-i),
// Zrec = [s_hor | s_ver] (64x1024 bf16) * WsPacked(1024x2560). Missing
// neighbours (grid edge) read a memset-zero 64x512 bf16 block -> branch-free,
// exact (zero A contributes zero). grid = (2560/256, 1, ncells).
// ---------------------------------------------------------------------------
__device__ __forceinline__ FragSet load_frags_dec(const __bf16* __restrict__ hor,
                                                  const __bf16* __restrict__ ver,
                                                  int m0, const __bf16* __restrict__ WsP,
                                                  int n0, int kk, int lane) {
    const __bf16* Ap = (kk < DD) ? hor : ver;
    int kl = (kk < DD) ? kk : kk - DD;
    FragSet f;
    f.a0 = load_a_frag(Ap, DD, m0, kl, lane);
    f.a1 = load_a_frag(Ap, DD, m0 + 16, kl, lane);
    f.b0 = *(const v16bf*)b_frag_ptr(WsP, 2 * DD, n0, kk, lane);
    f.b1 = *(const v16bf*)b_frag_ptr(WsP, 2 * DD, n0 + 16, kk, lane);
    f.b2 = *(const v16bf*)b_frag_ptr(WsP, 2 * DD, n0 + 32, kk, lane);
    f.b3 = *(const v16bf*)b_frag_ptr(WsP, 2 * DD, n0 + 48, kk, lane);
    return f;
}

__global__ void __launch_bounds__(256, 2)
gemm_dec(const __bf16* __restrict__ sprev, int i0, int d,
         const __bf16* __restrict__ WsP,
         const __bf16* __restrict__ zeroA,
         float* __restrict__ Zrec) {
    int lane = threadIdx.x & 31;
    int wave = threadIdx.x >> 5;
    int cell = blockIdx.z;
    int i = i0 + cell;
    int j = d - i;
    const __bf16* hor = (j >= 1) ? sprev + (size_t)i * (BSZ * DD) : zeroA;
    const __bf16* ver = (i >= 1) ? sprev + (size_t)(i - 1) * (BSZ * DD) : zeroA;
    int m0 = (wave & 1) * 32;
    int n0 = blockIdx.x * 256 + (wave >> 1) * 64;
    v8f acc[2][4] = {};

    FragSet f0 = load_frags_dec(hor, ver, m0, WsP, n0, 0, lane);
    for (int kk = 0; kk < 2 * DD; kk += 64) {
        FragSet f1 = load_frags_dec(hor, ver, m0, WsP, n0, kk + 32, lane);
        do_wmma(f0, acc);
        if (kk + 64 < 2 * DD) {
            f0 = load_frags_dec(hor, ver, m0, WsP, n0, kk + 64, lane);
            if (kk + 128 < 2 * DD)
                __builtin_prefetch(b_frag_ptr(WsP, 2 * DD, n0, kk + 128, lane), 0, 1);
        }
        do_wmma(f1, acc);
    }
    store_tile(acc, Zrec + (size_t)cell * (BSZ * 5 * DD), 5 * DD, m0, n0, nullptr, lane);
}

// ---------------------------------------------------------------------------
// Weight packer: fp32 source (arbitrary strides so both (N,K) "@W.T" and (K,N)
// "@W" layouts work) -> bf16 WMMA-B packed layout.
// ---------------------------------------------------------------------------
__global__ void pack_b(const float* __restrict__ W, __bf16* __restrict__ out,
                       int Kdim, int Ndim, long sk, long sn) {
    size_t id = (size_t)blockIdx.x * blockDim.x + threadIdx.x;
    size_t total = (size_t)Kdim * Ndim;
    if (id >= total) return;
    int e    = (int)(id & 15);
    int lane = (int)((id >> 4) & 31);
    size_t rest = id >> 9;
    int kt = (int)(rest % (size_t)(Kdim >> 5));
    int nt = (int)(rest / (size_t)(Kdim >> 5));
    int khalf = lane >> 4;
    int kloc  = (e < 8) ? (khalf * 8 + e) : (16 + khalf * 8 + (e - 8));
    int k = kt * 32 + kloc;
    int n = nt * 16 + (lane & 15);
    out[id] = (__bf16)W[(long)k * sk + (long)n * sn];
}

// Embedding gather (with optional teacher-forcing shift + BOS) -> bf16 rows
__global__ void embed_gather(const int* __restrict__ tok, const float* __restrict__ emb,
                             __bf16* __restrict__ out, int T, int Bn, int Ed,
                             int shift, int bos) {
    size_t id = (size_t)blockIdx.x * blockDim.x + threadIdx.x;
    size_t total = (size_t)T * Bn * Ed;
    if (id >= total) return;
    int e = (int)(id % Ed);
    int b = (int)((id / Ed) % Bn);
    int t = (int)(id / ((size_t)Ed * Bn));
    int v = (shift && t == 0) ? bos : tok[(size_t)(t - shift) * Bn + b];
    out[id] = (__bf16)emb[(size_t)v * Ed + e];
}

// Encoder LSTM step: z = Xp[t] (incl. bias) + h@Whh^T ; torch gate order i,f,g,o
__global__ void enc_step(const float* __restrict__ Xp_t, const float* __restrict__ Zh,
                         float* __restrict__ cst, __bf16* __restrict__ hst,
                         __bf16* __restrict__ h_all, int t, int dir_off) {
    int id = blockIdx.x * blockDim.x + threadIdx.x;   // 64*256
    if (id >= BSZ * DENC_) return;
    int dch = id & (DENC_ - 1);
    int b   = id >> 8;
    const float* zr = Xp_t + (size_t)b * (4 * DENC_);
    const float* hr = Zh   + (size_t)b * (4 * DENC_);
    float zi = zr[dch]             + hr[dch];
    float zf = zr[DENC_ + dch]     + hr[DENC_ + dch];
    float zg = zr[2 * DENC_ + dch] + hr[2 * DENC_ + dch];
    float zo = zr[3 * DENC_ + dch] + hr[3 * DENC_ + dch];
    float c = cst[id];
    c = sigm_(zf) * c + sigm_(zi) * tanhf(zg);
    float h = sigm_(zo) * tanhf(c);
    cst[id] = c;
    hst[id] = (__bf16)h;
    h_all[((size_t)t * BSZ + b) * (2 * DENC_) + dir_off + dch] = (__bf16)h;
}

// 2D-LSTM cell elementwise: gates i,f,o,l,g (5*D wide); c fp32, s bf16
__global__ void cell_ew(const float* __restrict__ Hx, const float* __restrict__ Yx,
                        const float* __restrict__ Zrec,
                        const float* __restrict__ cprev, float* __restrict__ ccur,
                        __bf16* __restrict__ scur, __bf16* __restrict__ SoutB,
                        int i0, int d) {
    size_t id = (size_t)blockIdx.x * blockDim.x + threadIdx.x;
    int n    = (int)(id & (DD - 1));
    int b    = (int)((id >> 9) & (BSZ - 1));
    int cell = (int)(id >> 15);
    int i = i0 + cell;
    int j = d - i;
    const float* hx = Hx   + ((size_t)i * BSZ + b) * (5 * DD);
    const float* yx = Yx   + ((size_t)j * BSZ + b) * (5 * DD);
    const float* zr = Zrec + ((size_t)cell * BSZ + b) * (5 * DD);
    float zi = hx[n]          + yx[n]          + zr[n];
    float zf = hx[DD + n]     + yx[DD + n]     + zr[DD + n];
    float zo = hx[2 * DD + n] + yx[2 * DD + n] + zr[2 * DD + n];
    float zl = hx[3 * DD + n] + yx[3 * DD + n] + zr[3 * DD + n];
    float zg = hx[4 * DD + n] + yx[4 * DD + n] + zr[4 * DD + n];
    size_t sidx = ((size_t)i * BSZ + b) * DD + n;
    float ch = (j >= 1) ? cprev[sidx] : 0.0f;
    float cv = (i >= 1) ? cprev[((size_t)(i - 1) * BSZ + b) * DD + n] : 0.0f;
    float l = sigm_(zl);
    float c = sigm_(zf) * (l * ch + (1.0f - l) * cv) + sigm_(zi) * tanhf(zg);
    float s = sigm_(zo) * tanhf(c);
    ccur[sidx] = c;
    scur[sidx] = (__bf16)s;
    if (i == LIN_ - 1)
        SoutB[((size_t)j * BSZ + b) * DD + n] = (__bf16)s;
}

// ---------------------------------------------------------------------------
static inline size_t al256(size_t x) { return (x + 255) & ~(size_t)255; }

extern "C" void kernel_launch(void* const* d_in, const int* in_sizes, int n_in,
                              void* d_out, int out_size, void* d_ws, size_t ws_size,
                              hipStream_t stream) {
    const int*   x      = (const int*)d_in[0];
    const int*   y      = (const int*)d_in[2];
    const float* emb_in = (const float*)d_in[4];
    const float* emb_out= (const float*)d_in[5];
    const float* Wih_f  = (const float*)d_in[6];
    const float* Whh_f  = (const float*)d_in[7];
    const float* b_f    = (const float*)d_in[8];
    const float* Wih_b  = (const float*)d_in[9];
    const float* Whh_b  = (const float*)d_in[10];
    const float* b_b    = (const float*)d_in[11];
    const float* Wx     = (const float*)d_in[12];
    const float* Ws     = (const float*)d_in[13];
    const float* b_cell = (const float*)d_in[14];
    const float* W_log  = (const float*)d_in[15];
    const float* b_log  = (const float*)d_in[16];
    float* out = (float*)d_out;

    // ---- workspace carve-out ----
    char* p = (char*)d_ws;
    auto grab = [&](size_t bytes) { char* r = p; p += al256(bytes); return (void*)r; };
    __bf16* pkWihF = (__bf16*)grab((size_t)E_ * 4 * DENC_ * 2);
    __bf16* pkWhhF = (__bf16*)grab((size_t)DENC_ * 4 * DENC_ * 2);
    __bf16* pkWihB = (__bf16*)grab((size_t)E_ * 4 * DENC_ * 2);
    __bf16* pkWhhB = (__bf16*)grab((size_t)DENC_ * 4 * DENC_ * 2);
    __bf16* pkWxT  = (__bf16*)grab((size_t)(2 * DENC_) * 5 * DD * 2);
    __bf16* pkWxB  = (__bf16*)grab((size_t)E_ * 5 * DD * 2);
    __bf16* pkWs   = (__bf16*)grab((size_t)(2 * DD) * 5 * DD * 2);
    __bf16* pkWlog = (__bf16*)grab((size_t)DD * VOUT_ * 2);
    __bf16* xemb   = (__bf16*)grab((size_t)LIN_ * BSZ * E_ * 2);
    __bf16* yemb   = (__bf16*)grab((size_t)LOUT_ * BSZ * E_ * 2);
    float*  XpF    = (float*)grab((size_t)LIN_ * BSZ * 4 * DENC_ * 4);
    float*  XpB    = (float*)grab((size_t)LIN_ * BSZ * 4 * DENC_ * 4);
    float*  zbuf   = (float*)grab((size_t)BSZ * 4 * DENC_ * 4);
    __bf16* hstate = (__bf16*)grab((size_t)BSZ * DENC_ * 2);
    float*  cstate = (float*)grab((size_t)BSZ * DENC_ * 4);
    __bf16* h_all  = (__bf16*)grab((size_t)LIN_ * BSZ * 2 * DENC_ * 2);
    float*  Hx     = (float*)grab((size_t)LIN_ * BSZ * 5 * DD * 4);
    float*  Yx     = (float*)grab((size_t)LOUT_ * BSZ * 5 * DD * 4);
    float*  Zrec   = (float*)grab((size_t)LIN_ * BSZ * 5 * DD * 4);
    __bf16* sbuf0  = (__bf16*)grab((size_t)LIN_ * BSZ * DD * 2);
    __bf16* sbuf1  = (__bf16*)grab((size_t)LIN_ * BSZ * DD * 2);
    float*  cbuf0  = (float*)grab((size_t)LIN_ * BSZ * DD * 4);
    float*  cbuf1  = (float*)grab((size_t)LIN_ * BSZ * DD * 4);
    __bf16* SoutB  = (__bf16*)grab((size_t)LOUT_ * BSZ * DD * 2);
    __bf16* zeroA  = (__bf16*)grab((size_t)BSZ * DD * 2);
    (void)ws_size; (void)n_in; (void)in_sizes; (void)out_size;

    auto packGrid = [](size_t tot) { return dim3((unsigned)((tot + 255) / 256)); };

    // ---- pack weights into WMMA-B layout (bf16) ----
    // x@W.T forms: B(k,n) = W[n*K + k]  -> sk=1, sn=K
    pack_b<<<packGrid((size_t)E_ * 4 * DENC_), 256, 0, stream>>>(Wih_f, pkWihF, E_, 4 * DENC_, 1, E_);
    pack_b<<<packGrid((size_t)DENC_ * 4 * DENC_), 256, 0, stream>>>(Whh_f, pkWhhF, DENC_, 4 * DENC_, 1, DENC_);
    pack_b<<<packGrid((size_t)E_ * 4 * DENC_), 256, 0, stream>>>(Wih_b, pkWihB, E_, 4 * DENC_, 1, E_);
    pack_b<<<packGrid((size_t)DENC_ * 4 * DENC_), 256, 0, stream>>>(Whh_b, pkWhhB, DENC_, 4 * DENC_, 1, DENC_);
    // x@W forms: B(k,n) = W[k*N + n]    -> sk=N, sn=1
    pack_b<<<packGrid((size_t)(2 * DENC_) * 5 * DD), 256, 0, stream>>>(Wx, pkWxT, 2 * DENC_, 5 * DD, 5 * DD, 1);
    pack_b<<<packGrid((size_t)E_ * 5 * DD), 256, 0, stream>>>(Wx + (size_t)(2 * DENC_) * 5 * DD, pkWxB, E_, 5 * DD, 5 * DD, 1);
    pack_b<<<packGrid((size_t)(2 * DD) * 5 * DD), 256, 0, stream>>>(Ws, pkWs, 2 * DD, 5 * DD, 5 * DD, 1);
    pack_b<<<packGrid((size_t)DD * VOUT_), 256, 0, stream>>>(W_log, pkWlog, DD, VOUT_, VOUT_, 1);

    // ---- embeddings + zero edge-block ----
    embed_gather<<<packGrid((size_t)LIN_ * BSZ * E_), 256, 0, stream>>>(x, emb_in, xemb, LIN_, BSZ, E_, 0, 0);
    embed_gather<<<packGrid((size_t)LOUT_ * BSZ * E_), 256, 0, stream>>>(y, emb_out, yemb, LOUT_, BSZ, E_, 1, 1);
    hipMemsetAsync(zeroA, 0, (size_t)BSZ * DD * 2, stream);

    // ---- encoder input projections (bias folded in) ----
    gemm_bf16<<<dim3(4 * DENC_ / 256, LIN_ * BSZ / 64), 256, 0, stream>>>(xemb, E_, pkWihF, b_f, XpF, 4 * DENC_, E_);
    gemm_bf16<<<dim3(4 * DENC_ / 256, LIN_ * BSZ / 64), 256, 0, stream>>>(xemb, E_, pkWihB, b_b, XpB, 4 * DENC_, E_);

    // ---- forward encoder recurrence ----
    hipMemsetAsync(hstate, 0, (size_t)BSZ * DENC_ * 2, stream);
    hipMemsetAsync(cstate, 0, (size_t)BSZ * DENC_ * 4, stream);
    for (int t = 0; t < LIN_; ++t) {
        gemm_bf16<<<dim3(4 * DENC_ / 256, 1), 256, 0, stream>>>(hstate, DENC_, pkWhhF, nullptr, zbuf, 4 * DENC_, DENC_);
        enc_step<<<BSZ, DENC_, 0, stream>>>(XpF + (size_t)t * BSZ * 4 * DENC_, zbuf, cstate, hstate, h_all, t, 0);
    }
    // ---- backward encoder recurrence ----
    hipMemsetAsync(hstate, 0, (size_t)BSZ * DENC_ * 2, stream);
    hipMemsetAsync(cstate, 0, (size_t)BSZ * DENC_ * 4, stream);
    for (int t = LIN_ - 1; t >= 0; --t) {
        gemm_bf16<<<dim3(4 * DENC_ / 256, 1), 256, 0, stream>>>(hstate, DENC_, pkWhhB, nullptr, zbuf, 4 * DENC_, DENC_);
        enc_step<<<BSZ, DENC_, 0, stream>>>(XpB + (size_t)t * BSZ * 4 * DENC_, zbuf, cstate, hstate, h_all, t, DENC_);
    }

    // ---- decoder input projections: Hx = h@Wx_top ; Yx = y@Wx_bot + b_cell ----
    gemm_bf16<<<dim3(5 * DD / 256, LIN_ * BSZ / 64), 256, 0, stream>>>(h_all, 2 * DENC_, pkWxT, nullptr, Hx, 5 * DD, 2 * DENC_);
    gemm_bf16<<<dim3(5 * DD / 256, LOUT_ * BSZ / 64), 256, 0, stream>>>(yemb, E_, pkWxB, b_cell, Yx, 5 * DD, E_);

    // ---- 2D-LSTM anti-diagonal wavefront ----
    __bf16* sb[2] = { sbuf0, sbuf1 };
    float*  cb[2] = { cbuf0, cbuf1 };
    for (int d = 0; d < LIN_ + LOUT_ - 1; ++d) {
        int i0 = d - (LOUT_ - 1); if (i0 < 0) i0 = 0;
        int i1 = (d < LIN_ - 1) ? d : LIN_ - 1;
        int nc = i1 - i0 + 1;
        int cur = d & 1, prv = cur ^ 1;
        gemm_dec<<<dim3(5 * DD / 256, 1, nc), 256, 0, stream>>>(sb[prv], i0, d, pkWs, zeroA, Zrec);
        cell_ew<<<dim3((unsigned)(nc * BSZ * DD / 256)), 256, 0, stream>>>(Hx, Yx, Zrec, cb[prv], cb[cur], sb[cur], SoutB, i0, d);
    }

    // ---- logits: (LOUT*B, D) @ (D, VOUT) + b_log -> d_out ----
    gemm_bf16<<<dim3(VOUT_ / 256, LOUT_ * BSZ / 64), 256, 0, stream>>>(SoutB, DD, pkWlog, b_log, out, VOUT_, DD);
}
